// GCNRes_82592221102191
// MI455X (gfx1250) — matrix operand order repaired and checked by reading, not compiled, not run on hardware
//
#include <hip/hip_runtime.h>
#include <hip/hip_bf16.h>
#include <stdint.h>

// ---------------------------------------------------------------------------
// GCN residual block for MI455X (gfx1250, wave32, WMMA)
//   h  = relu(BN1(scatter(x @ W1^T)))
//   o  = relu(BN2(scatter(h @ W2^T)) + x)
// GEMMs: bf16 inputs, f32 accumulate via v_wmma_f32_16x16x32_bf16.
// Scatter: L2-resident gather + f32 global atomics (self-loops folded into
// a deterministic non-atomic init pass).
// ---------------------------------------------------------------------------

typedef __attribute__((ext_vector_type(16))) __bf16      v16bf;
typedef __attribute__((ext_vector_type(8)))  float       v8f;
typedef __attribute__((ext_vector_type(4)))  unsigned int u32x4;

__device__ __forceinline__ unsigned short f32_to_bf16_rne(float f) {
  union { float f; unsigned u; } x; x.f = f;
  unsigned u = x.u;
  u += 0x7FFFu + ((u >> 16) & 1u);           // round-to-nearest-even
  return (unsigned short)(u >> 16);
}

// ---- elementwise f32 -> bf16 (stored as u16 bit pattern) -------------------
__global__ void k_cvt_bf16(const float* __restrict__ in,
                           unsigned short* __restrict__ out, int n) {
  for (int i = blockIdx.x * blockDim.x + threadIdx.x; i < n;
       i += gridDim.x * blockDim.x)
    out[i] = f32_to_bf16_rne(in[i]);
}

// ---- degree (in-degree over col, self-loop => init 1) ----------------------
__global__ void k_deg_init(float* __restrict__ deg, int n) {
  int i = blockIdx.x * blockDim.x + threadIdx.x;
  if (i < n) deg[i] = 1.0f;                   // self loop
}
__global__ void k_deg_accum(const int* __restrict__ col, float* __restrict__ deg,
                            int E, int N) {
  int e = blockIdx.x * blockDim.x + threadIdx.x;
  if (e < E) {
    unsigned c = (unsigned)col[e];
    if (c < (unsigned)N) atomicAdd(&deg[c], 1.0f);
  }
}
__global__ void k_dis(const float* __restrict__ deg, float* __restrict__ dis, int n) {
  int i = blockIdx.x * blockDim.x + threadIdx.x;
  if (i < n) dis[i] = rsqrtf(deg[i]);         // deg >= 1 always
}

// ---- WMMA GEMM: out[M,OC] = A[M,K](bf16) * W[OC,K](bf16)^T, f32 accumulate -
// One wave computes a 16(M) x 64(OC) tile: 4 accumulators sharing one A frag.
__global__ void k_wmma_gemm(const unsigned short* __restrict__ A,
                            const unsigned short* __restrict__ W,
                            float* __restrict__ out,
                            int M, int K, int OC) {
  const int lane = threadIdx.x & 31;
  const int wave = (blockIdx.x * blockDim.x + threadIdx.x) >> 5;
  const int nt64 = OC >> 6;
  const int totalTiles = (M >> 4) * nt64;
  if (wave >= totalTiles) return;             // wave-uniform: EXEC stays all-1s
  const int mt = wave / nt64;
  const int nt = wave % nt64;
  const int h  = lane >> 4;                   // lane half (K-phase select)
  const int ln = lane & 15;

  const unsigned short* arow = A + (size_t)(mt * 16 + ln) * K;
  const unsigned short* wrow = W + (size_t)(nt * 64 + ln) * K + h * 16;
  const size_t wstep = (size_t)16 * K;        // 16 output channels

  v8f acc0 = {}, acc1 = {}, acc2 = {}, acc3 = {};
  for (int k = 0; k < K; k += 32) {
    union Frag { v16bf v; u32x4 u[2]; } a, b0, b1, b2, b3;
    // A 16x32 bf16 layout: lane half h holds K = h*16..h*16+15 (contiguous, matches B)
    a.u[0] = *(const u32x4*)(arow + k + h * 16);
    a.u[1] = *(const u32x4*)(arow + k + h * 16 + 8);
    // B 32x16: column = W row (K-contiguous); lane half selects K 0-15/16-31
    const unsigned short* w = wrow + k;
    b0.u[0] = *(const u32x4*)(w);             b0.u[1] = *(const u32x4*)(w + 8);
    w += wstep;
    b1.u[0] = *(const u32x4*)(w);             b1.u[1] = *(const u32x4*)(w + 8);
    w += wstep;
    b2.u[0] = *(const u32x4*)(w);             b2.u[1] = *(const u32x4*)(w + 8);
    w += wstep;
    b3.u[0] = *(const u32x4*)(w);             b3.u[1] = *(const u32x4*)(w + 8);

    acc0 = __builtin_amdgcn_wmma_f32_16x16x32_bf16(false, a.v, false, b0.v,
                                                   (short)0, acc0, false, false);
    acc1 = __builtin_amdgcn_wmma_f32_16x16x32_bf16(false, a.v, false, b1.v,
                                                   (short)0, acc1, false, false);
    acc2 = __builtin_amdgcn_wmma_f32_16x16x32_bf16(false, a.v, false, b2.v,
                                                   (short)0, acc2, false, false);
    acc3 = __builtin_amdgcn_wmma_f32_16x16x32_bf16(false, a.v, false, b3.v,
                                                   (short)0, acc3, false, false);
  }
  // D layout: lane l, VGPR r -> row (l>>4)*8 + r, col l&15
  float* o = out + (size_t)(mt * 16 + h * 8) * OC + (size_t)nt * 64 + ln;
#pragma unroll
  for (int r = 0; r < 8; ++r) {
    o[(size_t)r * OC +  0] = acc0[r];
    o[(size_t)r * OC + 16] = acc1[r];
    o[(size_t)r * OC + 32] = acc2[r];
    o[(size_t)r * OC + 48] = acc3[r];
  }
}

// ---- agg init with self-loop contribution: agg = xl * dis[i]^2 -------------
__global__ void k_selfloop(const float* __restrict__ xl, const float* __restrict__ dis,
                           float* __restrict__ agg, int N, int C) {
  int total = N * C;
  for (int i = blockIdx.x * blockDim.x + threadIdx.x; i < total;
       i += gridDim.x * blockDim.x) {
    int node = i / C;
    float d = dis[node];
    agg[i] = xl[i] * d * d;
  }
}

// ---- edge scatter: agg[col] += xl[row] * dis[row]*dis[col] -----------------
__global__ void k_scatter(const float* __restrict__ xl, const int* __restrict__ row,
                          const int* __restrict__ col, const float* __restrict__ dis,
                          float* __restrict__ agg, int E, int C, int N) {
  const int per = C >> 2;                     // float4 chunks per edge
  const long long total = (long long)E * per;
  for (long long t = blockIdx.x * (long long)blockDim.x + threadIdx.x; t < total;
       t += (long long)gridDim.x * blockDim.x) {
    int e  = (int)(t / per);
    int c4 = ((int)(t % per)) << 2;
    unsigned r = (unsigned)row[e], c = (unsigned)col[e];
    if (r >= (unsigned)N || c >= (unsigned)N) continue;  // safety guard
    float nrm = dis[r] * dis[c];
    float4 v = *(const float4*)(xl + (size_t)r * C + c4); // L2-resident gather
    float* dst = agg + (size_t)c * C + c4;
    atomicAdd(dst + 0, v.x * nrm);
    atomicAdd(dst + 1, v.y * nrm);
    atomicAdd(dst + 2, v.z * nrm);
    atomicAdd(dst + 3, v.w * nrm);
  }
}

// ---- batchnorm statistics: one block per channel ---------------------------
__global__ void k_bn_stats(const float* __restrict__ h, int N, int C,
                           float* __restrict__ mean, float* __restrict__ invstd) {
  __shared__ float ss[256], sq[256];
  const int c = blockIdx.x;
  float s = 0.f, s2 = 0.f;
  for (int i = threadIdx.x; i < N; i += blockDim.x) {
    float v = h[(size_t)i * C + c];
    s += v; s2 += v * v;
  }
  ss[threadIdx.x] = s; sq[threadIdx.x] = s2; __syncthreads();
  for (int off = blockDim.x >> 1; off > 0; off >>= 1) {
    if (threadIdx.x < off) {
      ss[threadIdx.x] += ss[threadIdx.x + off];
      sq[threadIdx.x] += sq[threadIdx.x + off];
    }
    __syncthreads();
  }
  if (threadIdx.x == 0) {
    float m   = ss[0] / (float)N;
    float var = sq[0] / (float)N - m * m;     // biased, as BatchNorm1d training
    mean[c] = m;
    invstd[c] = rsqrtf(var + 1e-5f);
  }
}

// ---- BN + ReLU -> bf16 (feeds GEMM2) ---------------------------------------
__global__ void k_bn_relu_bf16(const float* __restrict__ agg, const float* __restrict__ mean,
                               const float* __restrict__ invstd, const float* __restrict__ g,
                               const float* __restrict__ b, unsigned short* __restrict__ out,
                               int N, int C) {
  int total = N * C;
  for (int i = blockIdx.x * blockDim.x + threadIdx.x; i < total;
       i += gridDim.x * blockDim.x) {
    int c = i % C;
    float v = (agg[i] - mean[c]) * invstd[c] * g[c] + b[c];
    out[i] = f32_to_bf16_rne(fmaxf(v, 0.f));
  }
}

// ---- BN + residual + ReLU -> f32 output ------------------------------------
__global__ void k_bn_res_relu(const float* __restrict__ agg, const float* __restrict__ mean,
                              const float* __restrict__ invstd, const float* __restrict__ g,
                              const float* __restrict__ b, const float* __restrict__ x,
                              float* __restrict__ out, int N, int C) {
  int total = N * C;
  for (int i = blockIdx.x * blockDim.x + threadIdx.x; i < total;
       i += gridDim.x * blockDim.x) {
    int c = i % C;
    float v = (agg[i] - mean[c]) * invstd[c] * g[c] + b[c] + x[i];
    out[i] = fmaxf(v, 0.f);
  }
}

// ---------------------------------------------------------------------------
extern "C" void kernel_launch(void* const* d_in, const int* in_sizes, int n_in,
                              void* d_out, int out_size, void* d_ws, size_t ws_size,
                              hipStream_t stream) {
  (void)in_sizes; (void)n_in; (void)out_size; (void)ws_size;
  const int N = 10000, C = 256, IC = 512, E = 160000;

  const float* x  = (const float*)d_in[0];
  const int*   es = (const int*)d_in[1];      // [2,E]: row = es[0..E), col = es[E..2E)
  const float* W1 = (const float*)d_in[2];
  const float* g1 = (const float*)d_in[3];
  const float* b1 = (const float*)d_in[4];
  const float* W2 = (const float*)d_in[5];
  const float* g2 = (const float*)d_in[6];
  const float* b2 = (const float*)d_in[7];
  const int* row = es;
  const int* col = es + E;
  float* out = (float*)d_out;

  // bump allocator over workspace (256B aligned)
  char* wsb = (char*)d_ws;
  size_t cur = 0;
  auto alloc = [&](size_t bytes) -> void* {
    void* p = wsb + cur;
    cur = (cur + bytes + 255) & ~(size_t)255;
    return p;
  };
  float*          deg   = (float*)alloc((size_t)N * 4);
  float*          dis   = (float*)alloc((size_t)N * 4);
  float*          mean1 = (float*)alloc((size_t)IC * 4);
  float*          istd1 = (float*)alloc((size_t)IC * 4);
  float*          mean2 = (float*)alloc((size_t)C * 4);
  float*          istd2 = (float*)alloc((size_t)C * 4);
  unsigned short* xbf   = (unsigned short*)alloc((size_t)N * C * 2);
  unsigned short* w1bf  = (unsigned short*)alloc((size_t)IC * C * 2);
  unsigned short* w2bf  = (unsigned short*)alloc((size_t)C * IC * 2);
  unsigned short* hbf   = (unsigned short*)alloc((size_t)N * IC * 2);
  float*          xl    = (float*)alloc((size_t)N * IC * 4);  // xl1, reused as xl2
  float*          agg   = (float*)alloc((size_t)N * IC * 4);  // agg1, reused as agg2

  const int T = 256;
  auto cdiv = [](long long a, long long b) { return (int)((a + b - 1) / b); };

  // --- precision prep: bf16 operands for the matrix pipe ---
  k_cvt_bf16<<<cdiv((long long)N * C, T), T, 0, stream>>>(x, xbf, N * C);
  k_cvt_bf16<<<cdiv((long long)IC * C, T), T, 0, stream>>>(W1, w1bf, IC * C);
  k_cvt_bf16<<<cdiv((long long)C * IC, T), T, 0, stream>>>(W2, w2bf, C * IC);

  // --- symmetric normalization coefficients ---
  k_deg_init<<<cdiv(N, T), T, 0, stream>>>(deg, N);
  k_deg_accum<<<cdiv(E, T), T, 0, stream>>>(col, deg, E, N);
  k_dis<<<cdiv(N, T), T, 0, stream>>>(deg, dis, N);

  // --- layer 1: GEMM (WMMA) + scatter + BN + ReLU ---
  {
    int waves = (N / 16) * (IC / 64);         // 5000
    k_wmma_gemm<<<cdiv(waves, 8), T, 0, stream>>>(xbf, w1bf, xl, N, C, IC);
  }
  k_selfloop<<<cdiv((long long)N * IC, T), T, 0, stream>>>(xl, dis, agg, N, IC);
  k_scatter<<<cdiv((long long)E * (IC / 4), T), T, 0, stream>>>(xl, row, col, dis,
                                                                agg, E, IC, N);
  k_bn_stats<<<IC, T, 0, stream>>>(agg, N, IC, mean1, istd1);
  k_bn_relu_bf16<<<cdiv((long long)N * IC, T), T, 0, stream>>>(agg, mean1, istd1,
                                                               g1, b1, hbf, N, IC);

  // --- layer 2: GEMM (WMMA) + scatter + BN + residual + ReLU ---
  {
    int waves = (N / 16) * (C / 64);          // 2500
    k_wmma_gemm<<<cdiv(waves, 8), T, 0, stream>>>(hbf, w2bf, xl, N, IC, C);
  }
  k_selfloop<<<cdiv((long long)N * C, T), T, 0, stream>>>(xl, dis, agg, N, C);
  k_scatter<<<cdiv((long long)E * (C / 4), T), T, 0, stream>>>(xl, row, col, dis,
                                                               agg, E, C, N);
  k_bn_stats<<<C, T, 0, stream>>>(agg, N, C, mean2, istd2);
  k_bn_res_relu<<<cdiv((long long)N * C, T), T, 0, stream>>>(agg, mean2, istd2,
                                                             g2, b2, x, out, N, C);
}
